// Qwen3NextGatedDeltaNet_62019327754715
// MI455X (gfx1250) — matrix operand order repaired and checked by reading, not compiled, not run on hardware
//
#include <hip/hip_runtime.h>
#include <hip/hip_bf16.h>
#include <math.h>

// ---------------- problem constants ----------------
#define SEQ   512
#define DMODEL 2048
#define HKN   8
#define HVN   16
#define DKD   128
#define DVD   128
#define KW    4
#define KEY_DIM 1024           // HK*DK
#define VAL_DIM 2048           // HV*DV
#define NQKVZ  6144            // 2*KEY_DIM + 2*VAL_DIM
#define NBA    32              // 2*HV
#define CONVD  4096            // 2*KEY_DIM + VAL_DIM
#define EPSF   1e-6f

typedef __attribute__((ext_vector_type(16))) __bf16 v16bf;
typedef __attribute__((ext_vector_type(8)))  float  v8f;

__device__ __forceinline__ float silu_f(float x) { return x / (1.f + expf(-x)); }

// =====================================================================
// GEMM  C[M,N] = A[M,K] * B[K,N]   (fp32 in/out, compensated-bf16 WMMA)
// Block tile 128x64, BK=32, 256 threads = 8 waves (4 Mwaves x 2 Nwaves),
// each wave owns a 32x32 tile = 4 f32 accumulators, 12 WMMA / K-step.
// =====================================================================
#define LDSP 40  // bf16 row stride: 80B rows -> 16B aligned, bank-skewed

__global__ __launch_bounds__(256, 1)
void gemm_bf16comp(const float* __restrict__ A, const float* __restrict__ B,
                   float* __restrict__ C, int M, int N, int Kd)
{
  __shared__ __bf16 Ah[128][LDSP];
  __shared__ __bf16 Al[128][LDSP];
  __shared__ __bf16 Bth[64][LDSP];   // transposed: [n][k]
  __shared__ __bf16 Btl[64][LDSP];

  const int tid  = threadIdx.x;
  const int wave = tid >> 5;
  const int lane = tid & 31;
  const int m0   = blockIdx.x * 128;
  const int n0   = blockIdx.y * 64;

  const int mw = wave & 3;           // 32-row slab within 128
  const int nw = wave >> 2;          // 32-col slab within 64

  v8f acc00 = {}, acc01 = {}, acc10 = {}, acc11 = {};   // [mi][ni]

  const int arow = tid >> 1, acol = (tid & 1) * 16;     // A: 128x32 coop load
  const int brow = tid >> 3, bcol = (tid & 7) * 8;      // B: 32x64 coop load

  const float* Ap = A + (long)(m0 + arow) * Kd + acol;
  const float* Bp = B + (long)brow * N + n0 + bcol;

  for (int k0 = 0; k0 < Kd; k0 += 32) {
    float ra[16], rb[8];
#pragma unroll
    for (int j = 0; j < 16; ++j) ra[j] = Ap[k0 + j];
#pragma unroll
    for (int j = 0; j < 8; ++j) rb[j] = Bp[(long)k0 * N + j];
    if (k0 + 32 < Kd) {                    // global_prefetch_b8 next K-slab
      __builtin_prefetch(Ap + k0 + 32, 0, 1);
      __builtin_prefetch(Bp + (long)(k0 + 32) * N, 0, 1);
    }
    __syncthreads();                       // prior iter's frag reads done
#pragma unroll
    for (int j = 0; j < 16; ++j) {
      __bf16 h = (__bf16)ra[j];
      Ah[arow][acol + j] = h;
      Al[arow][acol + j] = (__bf16)(ra[j] - (float)h);
    }
#pragma unroll
    for (int j = 0; j < 8; ++j) {
      __bf16 h = (__bf16)rb[j];
      Bth[bcol + j][brow] = h;
      Btl[bcol + j][brow] = (__bf16)(rb[j] - (float)h);
    }
    __syncthreads();

    const int hf = lane >> 4;
    const int ml = lane & 15;

    union Frag { v16bf v; uint4 q[2]; };
    Frag a0h, a0l, a1h, a1l, b0h, b0l, b1h, b1l;
    // A 16-bit layout: element e -> K = (e<8 ? e : e+8) + 8*half  => two b128 runs
    {
      const int r0 = mw * 32 + ml, r1 = r0 + 16;
      a0h.q[0] = *(const uint4*)&Ah[r0][8 * hf];
      a0h.q[1] = *(const uint4*)&Ah[r0][16 + 8 * hf];
      a0l.q[0] = *(const uint4*)&Al[r0][8 * hf];
      a0l.q[1] = *(const uint4*)&Al[r0][16 + 8 * hf];
      a1h.q[0] = *(const uint4*)&Ah[r1][8 * hf];
      a1h.q[1] = *(const uint4*)&Ah[r1][16 + 8 * hf];
      a1l.q[0] = *(const uint4*)&Al[r1][8 * hf];
      a1l.q[1] = *(const uint4*)&Al[r1][16 + 8 * hf];
    }
    // B 16-bit layout: element e -> K = (e<8 ? e : e+8) + 8*half  => two b128 runs
    {
      const int r0 = nw * 32 + ml, r1 = r0 + 16;
      b0h.q[0] = *(const uint4*)&Bth[r0][8 * hf];
      b0h.q[1] = *(const uint4*)&Bth[r0][16 + 8 * hf];
      b0l.q[0] = *(const uint4*)&Btl[r0][8 * hf];
      b0l.q[1] = *(const uint4*)&Btl[r0][16 + 8 * hf];
      b1h.q[0] = *(const uint4*)&Bth[r1][8 * hf];
      b1h.q[1] = *(const uint4*)&Bth[r1][16 + 8 * hf];
      b1l.q[0] = *(const uint4*)&Btl[r1][8 * hf];
      b1l.q[1] = *(const uint4*)&Btl[r1][16 + 8 * hf];
    }

    // D = Ah*Bh + Ah*Bl + Al*Bh  (compensated bf16 ~= fp32 accuracy)
    acc00 = __builtin_amdgcn_wmma_f32_16x16x32_bf16(false, a0h.v, false, b0h.v, (short)0, acc00, false, false);
    acc00 = __builtin_amdgcn_wmma_f32_16x16x32_bf16(false, a0h.v, false, b0l.v, (short)0, acc00, false, false);
    acc00 = __builtin_amdgcn_wmma_f32_16x16x32_bf16(false, a0l.v, false, b0h.v, (short)0, acc00, false, false);
    acc01 = __builtin_amdgcn_wmma_f32_16x16x32_bf16(false, a0h.v, false, b1h.v, (short)0, acc01, false, false);
    acc01 = __builtin_amdgcn_wmma_f32_16x16x32_bf16(false, a0h.v, false, b1l.v, (short)0, acc01, false, false);
    acc01 = __builtin_amdgcn_wmma_f32_16x16x32_bf16(false, a0l.v, false, b1h.v, (short)0, acc01, false, false);
    acc10 = __builtin_amdgcn_wmma_f32_16x16x32_bf16(false, a1h.v, false, b0h.v, (short)0, acc10, false, false);
    acc10 = __builtin_amdgcn_wmma_f32_16x16x32_bf16(false, a1h.v, false, b0l.v, (short)0, acc10, false, false);
    acc10 = __builtin_amdgcn_wmma_f32_16x16x32_bf16(false, a1l.v, false, b0h.v, (short)0, acc10, false, false);
    acc11 = __builtin_amdgcn_wmma_f32_16x16x32_bf16(false, a1h.v, false, b1h.v, (short)0, acc11, false, false);
    acc11 = __builtin_amdgcn_wmma_f32_16x16x32_bf16(false, a1h.v, false, b1l.v, (short)0, acc11, false, false);
    acc11 = __builtin_amdgcn_wmma_f32_16x16x32_bf16(false, a1l.v, false, b1h.v, (short)0, acc11, false, false);
  }

  // C/D layout: VGPR r, lane l: m = 8*(l>>4)+r (within 16-tile), n = l&15
  const int hf = lane >> 4;
  const int nl = lane & 15;
  float* Cp = C + (long)n0 + nw * 32 + nl;
#pragma unroll
  for (int mi = 0; mi < 2; ++mi) {
#pragma unroll
    for (int r = 0; r < 8; ++r) {
      const long m = m0 + mw * 32 + mi * 16 + 8 * hf + r;
      const v8f a0 = mi ? acc10 : acc00;
      const v8f a1 = mi ? acc11 : acc01;
      Cp[m * (long)N]      = a0[r];
      Cp[m * (long)N + 16] = a1[r];
    }
  }
}

// =====================================================================
// ba = H @ W_ba   (512 x 2048 x 32 — negligible)
// =====================================================================
__global__ __launch_bounds__(32)
void ba_kernel(const float* __restrict__ H, const float* __restrict__ Wba,
               float* __restrict__ ba)
{
  const int s = blockIdx.x, j = threadIdx.x;
  const float4* H4 = (const float4*)(H + (long)s * DMODEL);
  float acc = 0.f;
  for (int d4 = 0; d4 < DMODEL / 4; ++d4) {
    const float4 h = H4[d4];
    acc = fmaf(h.x, Wba[(long)(4 * d4 + 0) * NBA + j], acc);
    acc = fmaf(h.y, Wba[(long)(4 * d4 + 1) * NBA + j], acc);
    acc = fmaf(h.z, Wba[(long)(4 * d4 + 2) * NBA + j], acc);
    acc = fmaf(h.w, Wba[(long)(4 * d4 + 3) * NBA + j], acc);
  }
  ba[s * NBA + j] = acc;
}

// =====================================================================
// depthwise causal conv (K=4) + SiLU, gathering channels from qkvz layout
// =====================================================================
__device__ __forceinline__ int qkvz_col(int c) {
  if (c < KEY_DIM)            return (c >> 7) * 768 + (c & 127);                 // q
  if (c < 2 * KEY_DIM) { int cc = c - KEY_DIM;  return (cc >> 7) * 768 + 128 + (cc & 127); }  // k
  int cc = c - 2 * KEY_DIM;   return (cc >> 8) * 768 + 256 + (cc & 255);         // v
}

__global__ __launch_bounds__(256)
void conv_silu_kernel(const float* __restrict__ qkvz, const float* __restrict__ cw,
                      float* __restrict__ mixed)
{
  const int s = blockIdx.x;
  for (int c = threadIdx.x; c < CONVD; c += 256) {
    const int col = qkvz_col(c);
    float acc = 0.f;
#pragma unroll
    for (int t = 0; t < KW; ++t) {
      const int sp = s - (KW - 1) + t;
      const float xv = (sp >= 0) ? qkvz[(long)sp * NQKVZ + col] : 0.f;
      acc = fmaf(cw[c * KW + t], xv, acc);
    }
    mixed[(long)s * CONVD + c] = silu_f(acc);
  }
}

// =====================================================================
// prep: l2norm(q)*DK^-0.5, l2norm(k), repeat heads x2, fused kq row; beta, g
// =====================================================================
__global__ __launch_bounds__(256)
void prep_kernel(const float* __restrict__ mixed, const float* __restrict__ ba,
                 const float* __restrict__ dt_bias, const float* __restrict__ A_log,
                 float* __restrict__ kq, float* __restrict__ gbuf, float* __restrict__ bbuf)
{
  const int s = blockIdx.x;
  const int h = threadIdx.x >> 5;     // key head 0..7
  const int lane = threadIdx.x & 31;

  float qv[4], kv[4], sq = 0.f, sk = 0.f;
#pragma unroll
  for (int j = 0; j < 4; ++j) {
    const int d = lane + 32 * j;
    qv[j] = mixed[(long)s * CONVD + h * DKD + d];
    kv[j] = mixed[(long)s * CONVD + KEY_DIM + h * DKD + d];
    sq = fmaf(qv[j], qv[j], sq);
    sk = fmaf(kv[j], kv[j], sk);
  }
#pragma unroll
  for (int m = 16; m >= 1; m >>= 1) {
    sq += __shfl_xor(sq, m, 32);
    sk += __shfl_xor(sk, m, 32);
  }
  const float rq = rsqrtf(sq + EPSF) * 0.08838834764831845f;  // * DK^-0.5
  const float rk = rsqrtf(sk + EPSF);
#pragma unroll
  for (int j = 0; j < 4; ++j) {
    const int d = lane + 32 * j;
    const float qo = qv[j] * rq, ko = kv[j] * rk;
    // fused row: kq[s][hv][0:128]=k, [128:256]=q ; head-repeat R=2
    kq[((long)s * HVN + 2 * h    ) * 256 + d]       = ko;
    kq[((long)s * HVN + 2 * h + 1) * 256 + d]       = ko;
    kq[((long)s * HVN + 2 * h    ) * 256 + 128 + d] = qo;
    kq[((long)s * HVN + 2 * h + 1) * 256 + 128 + d] = qo;
  }

  if (threadIdx.x < HVN) {
    const int hv = threadIdx.x;
    const float b = ba[s * NBA + (hv >> 1) * 4 + (hv & 1)];
    const float a = ba[s * NBA + (hv >> 1) * 4 + 2 + (hv & 1)];
    const float x = a + dt_bias[hv];
    const float sp = (x > 20.f) ? x : log1pf(expf(x));  // softplus
    bbuf[s * HVN + hv] = 1.f / (1.f + expf(-b));
    gbuf[s * HVN + hv] = expf(-expf(A_log[hv]) * sp);
  }
}

// =====================================================================
// delta-rule scan: DV embarrassingly parallel -> 32 blocks (16 hv x 2 DV halves),
// 64 threads/block, one 128-deep state column per thread in VGPRs.
// kq rows staged with GLOBAL_LOAD_ASYNC_TO_LDS_B128 (ASYNCcnt), double-buffered
// so step s+1's 1KB stage overlaps step s's 512-FMA body.
// =====================================================================
__global__ __launch_bounds__(64)
void scan_kernel(const float* __restrict__ kq, const float* __restrict__ mixed,
                 const float* __restrict__ gbuf, const float* __restrict__ bbuf,
                 float* __restrict__ core)
{
  const int hv = blockIdx.x >> 1;
  const int dv = ((blockIdx.x & 1) * 64) + threadIdx.x;

  __shared__ __align__(16) float KQ[2][256];   // [slot][k(128) | q(128)]

  float S[DKD];
#pragma unroll
  for (int i = 0; i < DKD; ++i) S[i] = 0.f;

  // per-lane async copy: 64 lanes x 16B = one 1KB kq row -> LDS slot
#define ASYNC_STAGE(slot_, s_)                                                      \
  {                                                                                 \
    const float* src_ = kq + ((long)(s_) * HVN + hv) * 256 + 4 * threadIdx.x;       \
    unsigned dst_ = (unsigned)(unsigned long long)&KQ[(slot_)][4 * threadIdx.x];    \
    asm volatile("global_load_async_to_lds_b128 %0, %1, off"                        \
                 :: "v"(dst_), "v"(src_) : "memory");                               \
  }

  ASYNC_STAGE(0, 0)
  asm volatile("s_wait_asynccnt 0x0" ::: "memory");
  __syncthreads();

  for (int s = 0; s < SEQ; ++s) {
    const int buf = s & 1;
    if (s + 1 < SEQ) ASYNC_STAGE(buf ^ 1, s + 1)      // overlap next stage

    const float vv   = mixed[(long)s * CONVD + 2 * KEY_DIM + hv * DVD + dv];
    const float g    = gbuf[s * HVN + hv];
    const float beta = bbuf[s * HVN + hv];

    const float4* K4 = (const float4*)&KQ[buf][0];
    const float4* Q4 = (const float4*)&KQ[buf][128];

    float p0 = 0.f, p1 = 0.f, p2 = 0.f, p3 = 0.f;
#pragma unroll
    for (int i = 0; i < DKD / 4; ++i) {
      const float4 kk = K4[i];
      p0 = fmaf(kk.x, S[4 * i + 0], p0);
      p1 = fmaf(kk.y, S[4 * i + 1], p1);
      p2 = fmaf(kk.z, S[4 * i + 2], p2);
      p3 = fmaf(kk.w, S[4 * i + 3], p3);
    }
    const float pred  = ((p0 + p1) + (p2 + p3)) * g;   // k . (g*S_prev)
    const float delta = (vv - pred) * beta;

    float o0 = 0.f, o1 = 0.f, o2 = 0.f, o3 = 0.f;
#pragma unroll
    for (int i = 0; i < DKD / 4; ++i) {
      const float4 kk = K4[i];
      const float4 qq = Q4[i];
      const float s0 = fmaf(g, S[4 * i + 0], kk.x * delta);
      const float s1 = fmaf(g, S[4 * i + 1], kk.y * delta);
      const float s2 = fmaf(g, S[4 * i + 2], kk.z * delta);
      const float s3 = fmaf(g, S[4 * i + 3], kk.w * delta);
      S[4 * i + 0] = s0; S[4 * i + 1] = s1; S[4 * i + 2] = s2; S[4 * i + 3] = s3;
      o0 = fmaf(qq.x, s0, o0);
      o1 = fmaf(qq.y, s1, o1);
      o2 = fmaf(qq.z, s2, o2);
      o3 = fmaf(qq.w, s3, o3);
    }
    core[((long)s * HVN + hv) * DVD + dv] = (o0 + o1) + (o2 + o3);

    asm volatile("s_wait_asynccnt 0x0" ::: "memory");  // next slot landed
    __syncthreads();                                   // all lanes done with slots
  }
#undef ASYNC_STAGE
}

// =====================================================================
// RMS-norm over DV * norm_weight * silu(z)  (z read straight from qkvz)
// =====================================================================
__global__ __launch_bounds__(256)
void gatenorm_kernel(const float* __restrict__ core, const float* __restrict__ qkvz,
                     const float* __restrict__ nw, float* __restrict__ normed)
{
  const int s = blockIdx.x;
  const int wave = threadIdx.x >> 5;
  const int lane = threadIdx.x & 31;
#pragma unroll
  for (int hh = 0; hh < 2; ++hh) {
    const int hv = wave * 2 + hh;
    float cv[4], ss = 0.f;
#pragma unroll
    for (int j = 0; j < 4; ++j) {
      const int d = lane + 32 * j;
      cv[j] = core[((long)s * HVN + hv) * DVD + d];
      ss = fmaf(cv[j], cv[j], ss);
    }
#pragma unroll
    for (int m = 16; m >= 1; m >>= 1) ss += __shfl_xor(ss, m, 32);
    const float r = rsqrtf(ss * (1.f / DVD) + EPSF);
#pragma unroll
    for (int j = 0; j < 4; ++j) {
      const int d = lane + 32 * j;
      const float z = qkvz[(long)s * NQKVZ + (hv >> 1) * 768 + 512 + (hv & 1) * 128 + d];
      normed[(long)s * VAL_DIM + hv * DVD + d] = cv[j] * r * nw[d] * silu_f(z);
    }
  }
}

// =====================================================================
extern "C" void kernel_launch(void* const* d_in, const int* in_sizes, int n_in,
                              void* d_out, int out_size, void* d_ws, size_t ws_size,
                              hipStream_t stream)
{
  const float* H      = (const float*)d_in[0];  // [512,2048]
  const float* Wqkvz  = (const float*)d_in[1];  // [2048,6144]
  const float* Wba    = (const float*)d_in[2];  // [2048,32]
  const float* convw  = (const float*)d_in[3];  // [4096,1,4]
  const float* dtb    = (const float*)d_in[4];  // [16]
  const float* Alog   = (const float*)d_in[5];  // [16]
  const float* nw     = (const float*)d_in[6];  // [128]
  const float* Wout   = (const float*)d_in[7];  // [2048,2048]
  float* out = (float*)d_out;                   // [512,2048]

  float* ws    = (float*)d_ws;
  float* qkvz  = ws;                              // 512*6144
  float* ba    = qkvz + (long)SEQ * NQKVZ;        // 512*32
  float* mixed = ba + (long)SEQ * NBA;            // 512*4096
  float* kq    = mixed + (long)SEQ * CONVD;       // 512*16*256 (k||q fused rows)
  float* gb    = kq + (long)SEQ * HVN * 256;      // 512*16
  float* bb    = gb + (long)SEQ * HVN;            // 512*16
  float* corep = bb + (long)SEQ * HVN;            // 512*16*128
  float* normed = kq;  // overlay: kq dead after scan, reused for gated output

  // 1) qkvz GEMM (WMMA, compensated bf16)
  gemm_bf16comp<<<dim3(SEQ / 128, NQKVZ / 64), 256, 0, stream>>>(H, Wqkvz, qkvz, SEQ, NQKVZ, DMODEL);
  // 2) ba GEMM (tiny)
  ba_kernel<<<dim3(SEQ), 32, 0, stream>>>(H, Wba, ba);
  // 3) conv + SiLU
  conv_silu_kernel<<<dim3(SEQ), 256, 0, stream>>>(qkvz, convw, mixed);
  // 4) l2norm / repeat / gates
  prep_kernel<<<dim3(SEQ), 256, 0, stream>>>(mixed, ba, dtb, Alog, kq, gb, bb);
  // 5) delta-rule scan (DV-parallel, register state, async-LDS staging)
  scan_kernel<<<dim3(HVN * 2), 64, 0, stream>>>(kq, mixed, gb, bb, corep);
  // 6) RMS-norm + gate
  gatenorm_kernel<<<dim3(SEQ), 256, 0, stream>>>(corep, qkvz, nw, normed);
  // 7) output GEMM (WMMA)
  gemm_bf16comp<<<dim3(SEQ / 128, DMODEL / 64), 256, 0, stream>>>(normed, Wout, out, SEQ, DMODEL, DMODEL);
}
